// BipartiteGNN_46669114638612
// MI455X (gfx1250) — compile-verified, matrix operand order checked by reading, not run on hardware
//
#include <hip/hip_runtime.h>
#include <hip/hip_bf16.h>

// Problem constants (match reference)
#define V_N     100000
#define E_N     200000
#define VOCAB_N 1000
#define D_N     128
#define ITERS_N 6
#define NNZ_N   (2 * E_N)
#define FOURD   512     // 4*D
#define KCAT    256     // D (x) + D (h)

typedef __attribute__((ext_vector_type(16))) __bf16 v16bf;
typedef __attribute__((ext_vector_type(8)))  float  v8f;

union Frag16 {
    v16bf v;
    unsigned u[8];
};

__device__ __forceinline__ unsigned short f2bf(float x) {
    unsigned u = __float_as_uint(x);
    u += 0x7FFFu + ((u >> 16) & 1u);   // round-to-nearest-even
    return (unsigned short)(u >> 16);
}

__device__ __forceinline__ float sigmoidf_(float x) {
    return 1.0f / (1.0f + __expf(-x));
}

// ---------------------------------------------------------------------------
// Utility: zero a float buffer (grid-stride)
// ---------------------------------------------------------------------------
__global__ void k_zero(float* __restrict__ p, long n) {
    long i = (long)blockIdx.x * blockDim.x + threadIdx.x;
    long stride = (long)gridDim.x * blockDim.x;
    for (; i < n; i += stride) p[i] = 0.0f;
}

// ---------------------------------------------------------------------------
// Pack [Wih;Whh] (each [512,128] f32) into bf16 B-fragment order:
// out[((kk*32 + nblk)*32 + lane)*16 + e], kk: K-step of 32, nblk: 16-col block.
// B[k][n] = Wcat[n][k]; within a lane, element e=(j,hi) maps to
// kloc = (j<4 ? half*8+2j : 16+half*8+2(j-4)) + hi, half = lane>>4 (ISA layout).
// Total 8*32*32*16 = 131072 elements.
// ---------------------------------------------------------------------------
__global__ void k_prep_pack_lstm(const float* __restrict__ Wih,
                                 const float* __restrict__ Whh,
                                 unsigned short* __restrict__ out) {
    int tid  = blockIdx.x * blockDim.x + threadIdx.x;   // 0..131071
    int e    = tid & 15;
    int lane = (tid >> 4) & 31;
    int nblk = (tid >> 9) & 31;
    int kk   = tid >> 14;
    int j = e >> 1, hi = e & 1;
    int half = lane >> 4;
    int kloc = (j < 4) ? (half * 8 + 2 * j + hi)
                       : (16 + half * 8 + 2 * (j - 4) + hi);
    int k = kk * 32 + kloc;
    int n = nblk * 16 + (lane & 15);
    float v = (k < D_N) ? Wih[n * D_N + k] : Whh[n * D_N + (k - D_N)];
    out[tid] = f2bf(v);
}

// Pack Wout^T ([1000,128] f32 -> K=128, N padded to 1024) in fragment order.
// out size: 4*64*32*16 = 131072 elements.
__global__ void k_prep_pack_wout(const float* __restrict__ Wout,
                                 unsigned short* __restrict__ out) {
    int tid  = blockIdx.x * blockDim.x + threadIdx.x;   // 0..131071
    int e    = tid & 15;
    int lane = (tid >> 4) & 31;
    int nblk = (tid >> 9) & 63;
    int kk   = tid >> 15;
    int j = e >> 1, hi = e & 1;
    int half = lane >> 4;
    int kloc = (j < 4) ? (half * 8 + 2 * j + hi)
                       : (16 + half * 8 + 2 * (j - 4) + hi);
    int k = kk * 32 + kloc;
    int n = nblk * 16 + (lane & 15);
    float v = (n < VOCAB_N) ? Wout[n * D_N + k] : 0.0f;
    out[tid] = f2bf(v);
}

__global__ void k_prep_bias(const float* __restrict__ bih,
                            const float* __restrict__ bhh,
                            float* __restrict__ b) {
    int i = blockIdx.x * blockDim.x + threadIdx.x;
    if (i < FOURD) b[i] = bih[i] + bhh[i];
}

__global__ void k_prep_edgevec(const float* __restrict__ w,
                               const float* __restrict__ b,
                               float* __restrict__ ev) {
    int d = threadIdx.x;
    if (d < D_N) ev[d] = w[d] + b[d];    // Linear(1,D) applied to ones(1)
}

// ---------------------------------------------------------------------------
// State init
// ---------------------------------------------------------------------------
__global__ void k_init_vert(const int* __restrict__ xv,
                            const float* __restrict__ emb,
                            float* __restrict__ h, float* __restrict__ c) {
    long i = (long)blockIdx.x * blockDim.x + threadIdx.x;   // V*D threads
    int row = (int)(i >> 7);
    int d   = (int)(i & 127);
    int x = xv[row];
    if (x == -1) x = VOCAB_N;            // padding idx
    h[i] = emb[(long)x * D_N + d];
    c[i] = 0.0f;
}

__global__ void k_init_edge(const float* __restrict__ ev,
                            float* __restrict__ h, float* __restrict__ c) {
    long i = (long)blockIdx.x * blockDim.x + threadIdx.x;   // E*D threads
    h[i] = ev[(int)(i & 127)];
    c[i] = 0.0f;
}

// ---------------------------------------------------------------------------
// Segment-sum scatter: msg[dsti[nz]] += src[srci[nz]].  One wave32 per nnz,
// 4 floats per lane (D=128). global_atomic_add_f32.
// ---------------------------------------------------------------------------
__global__ void k_scatter(const int* __restrict__ dsti,
                          const int* __restrict__ srci,
                          const float* __restrict__ src,
                          float* __restrict__ msg) {
    int t = blockIdx.x * blockDim.x + threadIdx.x;
    int nz = t >> 5;
    int lane = t & 31;
    if (nz >= NNZ_N) return;
    long db = (long)dsti[nz] * D_N;
    long sb = (long)srci[nz] * D_N;
#pragma unroll
    for (int j = 0; j < 4; ++j) {
        int d = lane + 32 * j;
        atomicAdd(&msg[db + d], src[sb + d]);
    }
}

// ---------------------------------------------------------------------------
// Fused LSTM step:  z = [X|H] @ Wcat^T + b;  gates;  H,C updated in place.
// Block: 256 threads (8 waves) = 32 rows x 512 gate cols.
// Wave w owns N-blocks {w, w+8, w+16, w+24}  ->  its 4 accumulators per
// M-tile are exactly gates (i,f,g,o) for columns [16w,16w+16): gate math
// runs fully in registers off the WMMA accumulators (no Z round-trip).
// Each B fragment feeds two WMMAs (M-tiles 0 and 1).
// ---------------------------------------------------------------------------
__global__ void __launch_bounds__(256)
k_lstm(const float* __restrict__ X,         // [N,128] messages
       float* __restrict__ H,               // [N,128] hidden (in/out)
       float* __restrict__ C,               // [N,128] cell (in/out)
       const unsigned short* __restrict__ WbP,  // packed bf16 B operand
       const float* __restrict__ bias) {        // [512] bih+bhh
    __shared__ unsigned short sA[32 * KCAT];    // 16 KB bf16 [x|h] tile

    const int tid  = threadIdx.x;
    const int row0 = blockIdx.x * 32;

    // Stage [X|H] rows as bf16 into LDS (row-major, 256 K per row).
#pragma unroll
    for (int j = 0; j < 16; ++j) {
        int e = tid + 256 * j;               // 0..4095
        int r = e >> 7;
        int k = e & 127;
        long g = (long)(row0 + r) * D_N + k;
        sA[r * KCAT + k]       = f2bf(X[g]);
        sA[r * KCAT + 128 + k] = f2bf(H[g]);
    }
    __syncthreads();

    const int lane = tid & 31;
    const int wave = tid >> 5;
    const int half = lane >> 4;
    const int mr   = lane & 15;

    v8f acc[2][4] = {};

#pragma unroll
    for (int kk = 0; kk < 8; ++kk) {
        // A fragments (two M-tiles) from LDS: K pairs contiguous -> b32/b128.
        Frag16 a[2];
#pragma unroll
        for (int mt = 0; mt < 2; ++mt) {
            const unsigned short* ar = &sA[(mt * 16 + mr) * KCAT + kk * 32];
#pragma unroll
            for (int j = 0; j < 8; ++j) {
                int kloc = (j < 4) ? (half * 8 + 2 * j)
                                   : (16 + half * 8 + 2 * (j - 4));
                a[mt].u[j] = *(const unsigned*)(ar + kloc);
            }
        }
#pragma unroll
        for (int t = 0; t < 4; ++t) {
            int nblk = wave + t * 8;         // gate t, columns [16w,16w+16)
            Frag16 b;
            b.v = *reinterpret_cast<const v16bf*>(
                WbP + ((size_t)((kk * 32 + nblk) * 32 + lane) << 4));
            acc[0][t] = __builtin_amdgcn_wmma_f32_16x16x32_bf16(
                false, a[0].v, false, b.v, (short)0, acc[0][t], false, false);
            acc[1][t] = __builtin_amdgcn_wmma_f32_16x16x32_bf16(
                false, a[1].v, false, b.v, (short)0, acc[1][t], false, false);
        }
    }

    // Gate math in registers.  C/D layout: lane L, elem v ->
    // row = v + 8*(L>>4), col = 16*wave + (L&15) = hidden dim d.
    const int d = wave * 16 + mr;
    const float bi = bias[d];
    const float bf_ = bias[128 + d];
    const float bg = bias[256 + d];
    const float bo = bias[384 + d];
#pragma unroll
    for (int mt = 0; mt < 2; ++mt) {
#pragma unroll
        for (int v = 0; v < 8; ++v) {
            int r = row0 + mt * 16 + v + 8 * half;
            long idx = (long)r * D_N + d;
            float ig = sigmoidf_(acc[mt][0][v] + bi);
            float fg = sigmoidf_(acc[mt][1][v] + bf_);
            float gg = tanhf(acc[mt][2][v] + bg);
            float og = sigmoidf_(acc[mt][3][v] + bo);
            float cn = fg * C[idx] + ig * gg;
            C[idx] = cn;
            H[idx] = og * tanhf(cn);
        }
    }
}

// ---------------------------------------------------------------------------
// logits = h_v @ Wout^T + bout.  Grid (V/32, 8); block 256 (8 waves);
// each wave owns one 16-col block for two M-tiles, K=128 -> 4 WMMA steps,
// each B fragment feeds two WMMAs.
// ---------------------------------------------------------------------------
__global__ void __launch_bounds__(256)
k_logits(const float* __restrict__ Hv,
         const unsigned short* __restrict__ WoP,
         const float* __restrict__ bout,
         float* __restrict__ out) {
    __shared__ unsigned short sA[32 * D_N];  // 8 KB bf16 row tile

    const int tid  = threadIdx.x;
    const int row0 = blockIdx.x * 32;

#pragma unroll
    for (int j = 0; j < 16; ++j) {
        int e = tid + 256 * j;               // 0..4095
        int r = e >> 7;
        int k = e & 127;
        sA[r * D_N + k] = f2bf(Hv[(long)(row0 + r) * D_N + k]);
    }
    __syncthreads();

    const int lane = tid & 31;
    const int wave = tid >> 5;
    const int half = lane >> 4;
    const int mr   = lane & 15;
    const int nblk = blockIdx.y * 8 + wave;  // 0..63 (N padded to 1024)

    v8f acc[2] = {};

#pragma unroll
    for (int kk = 0; kk < 4; ++kk) {
        Frag16 a[2];
#pragma unroll
        for (int mt = 0; mt < 2; ++mt) {
            const unsigned short* ar = &sA[(mt * 16 + mr) * D_N + kk * 32];
#pragma unroll
            for (int j = 0; j < 8; ++j) {
                int kloc = (j < 4) ? (half * 8 + 2 * j)
                                   : (16 + half * 8 + 2 * (j - 4));
                a[mt].u[j] = *(const unsigned*)(ar + kloc);
            }
        }
        Frag16 b;
        b.v = *reinterpret_cast<const v16bf*>(
            WoP + ((size_t)((kk * 64 + nblk) * 32 + lane) << 4));
        acc[0] = __builtin_amdgcn_wmma_f32_16x16x32_bf16(
            false, a[0].v, false, b.v, (short)0, acc[0], false, false);
        acc[1] = __builtin_amdgcn_wmma_f32_16x16x32_bf16(
            false, a[1].v, false, b.v, (short)0, acc[1], false, false);
    }

    int col = nblk * 16 + mr;
    if (col < VOCAB_N) {
        float bo = bout[col];
#pragma unroll
        for (int mt = 0; mt < 2; ++mt) {
#pragma unroll
            for (int v = 0; v < 8; ++v) {
                int r = row0 + mt * 16 + v + 8 * half;
                out[(long)r * VOCAB_N + col] = acc[mt][v] + bo;
            }
        }
    }
}

// ---------------------------------------------------------------------------
// Host-side launch: carve workspace, prep weights, init state, 6 message-
// passing iterations, final logits.  All launches on `stream` (graph-safe).
// ---------------------------------------------------------------------------
extern "C" void kernel_launch(void* const* d_in, const int* in_sizes, int n_in,
                              void* d_out, int out_size, void* d_ws, size_t ws_size,
                              hipStream_t stream) {
    (void)in_sizes; (void)n_in; (void)out_size; (void)ws_size;

    const int*   x_v      = (const int*)d_in[0];
    const int*   adj_rows = (const int*)d_in[1];
    const int*   adj_cols = (const int*)d_in[2];
    const float* emb      = (const float*)d_in[3];
    const float* e_w      = (const float*)d_in[4];
    const float* e_b      = (const float*)d_in[5];
    const float* Wih_e    = (const float*)d_in[6];
    const float* Whh_e    = (const float*)d_in[7];
    const float* bih_e    = (const float*)d_in[8];
    const float* bhh_e    = (const float*)d_in[9];
    const float* Wih_v    = (const float*)d_in[10];
    const float* Whh_v    = (const float*)d_in[11];
    const float* bih_v    = (const float*)d_in[12];
    const float* bhh_v    = (const float*)d_in[13];
    const float* Wout     = (const float*)d_in[14];
    const float* bout     = (const float*)d_in[15];
    float* logits = (float*)d_out;

    // Workspace carve (256B aligned slices)
    char* ws = (char*)d_ws;
    size_t off = 0;
    auto carve = [&](size_t bytes) -> void* {
        void* p = ws + off;
        off += (bytes + 255) & ~(size_t)255;
        return p;
    };
    float* h_v   = (float*)carve((size_t)V_N * D_N * sizeof(float));
    float* c_v   = (float*)carve((size_t)V_N * D_N * sizeof(float));
    float* h_e   = (float*)carve((size_t)E_N * D_N * sizeof(float));
    float* c_e   = (float*)carve((size_t)E_N * D_N * sizeof(float));
    float* msg_e = (float*)carve((size_t)E_N * D_N * sizeof(float));
    float* msg_v = (float*)carve((size_t)V_N * D_N * sizeof(float));
    unsigned short* WbP_e = (unsigned short*)carve((size_t)131072 * 2);
    unsigned short* WbP_v = (unsigned short*)carve((size_t)131072 * 2);
    unsigned short* WoP   = (unsigned short*)carve((size_t)131072 * 2);
    float* bias_e = (float*)carve(FOURD * sizeof(float));
    float* bias_v = (float*)carve(FOURD * sizeof(float));
    float* evec   = (float*)carve(D_N * sizeof(float));

    // --- weight prep ---
    k_prep_pack_lstm<<<512, 256, 0, stream>>>(Wih_e, Whh_e, WbP_e);
    k_prep_pack_lstm<<<512, 256, 0, stream>>>(Wih_v, Whh_v, WbP_v);
    k_prep_pack_wout<<<512, 256, 0, stream>>>(Wout, WoP);
    k_prep_bias<<<2, 256, 0, stream>>>(bih_e, bhh_e, bias_e);
    k_prep_bias<<<2, 256, 0, stream>>>(bih_v, bhh_v, bias_v);
    k_prep_edgevec<<<1, 128, 0, stream>>>(e_w, e_b, evec);

    // --- state init ---
    k_init_vert<<<(V_N * D_N) / 256, 256, 0, stream>>>(x_v, emb, h_v, c_v);
    k_init_edge<<<(E_N * D_N) / 256, 256, 0, stream>>>(evec, h_e, c_e);

    // --- message passing iterations ---
    const int scat_blocks = (NNZ_N * 32) / 256;   // one wave per nnz
    for (int it = 0; it < ITERS_N; ++it) {
        // vertex -> edge
        k_zero<<<4096, 256, 0, stream>>>(msg_e, (long)E_N * D_N);
        k_scatter<<<scat_blocks, 256, 0, stream>>>(adj_rows, adj_cols, h_v, msg_e);
        k_lstm<<<E_N / 32, 256, 0, stream>>>(msg_e, h_e, c_e, WbP_e, bias_e);
        // edge -> vertex
        k_zero<<<4096, 256, 0, stream>>>(msg_v, (long)V_N * D_N);
        k_scatter<<<scat_blocks, 256, 0, stream>>>(adj_cols, adj_rows, h_e, msg_v);
        k_lstm<<<V_N / 32, 256, 0, stream>>>(msg_v, h_v, c_v, WbP_v, bias_v);
    }

    // --- output projection ---
    k_logits<<<dim3(V_N / 32, 8), 256, 0, stream>>>(h_v, WoP, bout, logits);
}
